// MoEFFN_11441792877030
// MI455X (gfx1250) — matrix operand order repaired and checked by reading, not compiled, not run on hardware
//
#include <hip/hip_runtime.h>
#include <hip/hip_bf16.h>
#include <math.h>

// MoE FFN, top-2 of 8 experts.  B*S = 8192 tokens, D = 1024, F = 4096.
#define NE   8
#define DD   1024
#define FF   4096
#define TT   8192
#define TM   32          // tokens per FFN block
#define XS_STRIDE 1032   // bf16 elems; 2064B row -> bank-conflict-free b128 reads
#define HS_STRIDE 136    // bf16 elems; 272B row

typedef __attribute__((ext_vector_type(16))) __bf16 v16bf;
typedef __attribute__((ext_vector_type(4)))  __bf16 v4bf;
typedef __attribute__((ext_vector_type(8)))  float  v8f;
typedef __attribute__((ext_vector_type(4)))  unsigned int v4u;
typedef __attribute__((ext_vector_type(8)))  int v8i;
typedef __attribute__((ext_vector_type(4)))  int v4i;

// ---------------------------------------------------------------------------
// WMMA helper (CDNA5 wave32, 16x16x32 bf16 -> f32)
// ---------------------------------------------------------------------------
__device__ __forceinline__ v8f wmma_bf16(v16bf a, v16bf b, v8f c) {
    return __builtin_amdgcn_wmma_f32_16x16x32_bf16(false, a, false, b,
                                                   (short)0, c, false, false);
}

// A-fragment (16x32 bf16) from row-major LDS.
// lanes 0-15: elems 0..15 = K[0..15]
// lanes 16-31: elems 0..15 = K[16..31]
template <int STRIDE>
__device__ __forceinline__ v16bf load_A_frag(const __bf16* lds, int k0, int lane) {
    const int m = lane & 15;
    const int s = (lane >> 4) * 16;
    const __bf16* p0 = lds + m * STRIDE + k0 + s;
    v16bf a;
#pragma unroll
    for (int j = 0; j < 8; ++j) { a[j] = p0[j]; a[8 + j] = p0[8 + j]; }
    return a;
}

// B-fragment (32x16): B[k][n] = W[nbase+n][kk+k], W row-major bf16 (pre-cvt).
__device__ __forceinline__ v16bf load_B_frag(const __bf16* __restrict__ W,
                                             long long row_stride,
                                             int nbase, int kk, int lane) {
    const int n  = lane & 15;
    const int kb = (lane >> 4) * 16;
    return *(const v16bf*)(W + (long long)(nbase + n) * row_stride + kk + kb);
}

// ---------------------------------------------------------------------------
// 0) zero output
// ---------------------------------------------------------------------------
__global__ void moe_zero_kernel(float4* __restrict__ p, int n4) {
    int i = blockIdx.x * blockDim.x + threadIdx.x;
    if (i < n4) p[i] = make_float4(0.f, 0.f, 0.f, 0.f);
}

// ---------------------------------------------------------------------------
// 0b) fp32 -> bf16 pre-conversion (weights and x), one-shot per call
// ---------------------------------------------------------------------------
__global__ void moe_cvt_kernel(const float4* __restrict__ src,
                               v4bf* __restrict__ dst, int n4) {
    int i = blockIdx.x * blockDim.x + threadIdx.x;
    if (i < n4) {
        float4 v = src[i];
        v4bf o;
        o[0] = (__bf16)v.x; o[1] = (__bf16)v.y;
        o[2] = (__bf16)v.z; o[3] = (__bf16)v.w;
        dst[i] = o;
    }
}

// ---------------------------------------------------------------------------
// 1) router: one wave per token, softmax -> top-2 -> renormalize
// ---------------------------------------------------------------------------
__global__ void moe_router_kernel(const float* __restrict__ x,
                                  const float* __restrict__ Wgate,
                                  float* __restrict__ wtok) {
    const int tid  = threadIdx.x;
    const int lane = tid & 31;
    const int t    = blockIdx.x * 8 + (tid >> 5);

    float acc[NE];
#pragma unroll
    for (int e = 0; e < NE; ++e) acc[e] = 0.f;

    const float* xr = x + (long long)t * DD;
    for (int j = lane; j < DD; j += 32) {
        float xv = xr[j];
#pragma unroll
        for (int e = 0; e < NE; ++e) acc[e] += xv * Wgate[e * DD + j];
    }
#pragma unroll
    for (int e = 0; e < NE; ++e) {
#pragma unroll
        for (int off = 16; off > 0; off >>= 1)
            acc[e] += __shfl_xor(acc[e], off, 32);
    }
    if (lane == 0) {
        float mx = acc[0];
#pragma unroll
        for (int e = 1; e < NE; ++e) mx = fmaxf(mx, acc[e]);
        float p[NE]; float s = 0.f;
#pragma unroll
        for (int e = 0; e < NE; ++e) { p[e] = __expf(acc[e] - mx); s += p[e]; }
        float inv = 1.f / s;
#pragma unroll
        for (int e = 0; e < NE; ++e) p[e] *= inv;
        int i1 = 0;
#pragma unroll
        for (int e = 1; e < NE; ++e) if (p[e] > p[i1]) i1 = e;
        int i2 = (i1 == 0) ? 1 : 0;
#pragma unroll
        for (int e = 0; e < NE; ++e) if (e != i1 && p[e] > p[i2]) i2 = e;
        float ws = p[i1] + p[i2];
#pragma unroll
        for (int e = 0; e < NE; ++e) wtok[t * NE + e] = 0.f;
        wtok[t * NE + i1] = p[i1] / ws;
        wtok[t * NE + i2] = p[i2] / ws;
    }
}

// ---------------------------------------------------------------------------
// 2) deterministic per-expert token compaction (one wave per expert)
// ---------------------------------------------------------------------------
__global__ void moe_compact_kernel(const float* __restrict__ wtok,
                                   int* __restrict__ idx,
                                   int* __restrict__ cnt) {
    const int e    = blockIdx.x;
    const int lane = threadIdx.x;
    int base = 0;
    for (int t0 = 0; t0 < TT; t0 += 32) {
        int t = t0 + lane;
        bool sel = wtok[t * NE + e] > 0.f;
        unsigned mask  = (unsigned)__ballot(sel);
        unsigned below = mask & ((1u << lane) - 1u);
        if (sel) idx[e * TT + base + __builtin_popcount(below)] = t;
        base += __builtin_popcount(mask);
    }
    if (lane == 0) cnt[e] = base;
}

// ---------------------------------------------------------------------------
// 3) grouped FFN: block = (32-token tile, expert), 8 waves.
//    x tile staged by TDM (tensor_load_to_lds), all math on bf16 WMMA.
// ---------------------------------------------------------------------------
__launch_bounds__(256, 1)
__global__ void moe_ffn_kernel(const __bf16* __restrict__ xb,
                               const __bf16* __restrict__ Wgb,
                               const __bf16* __restrict__ Wub,
                               const __bf16* __restrict__ Wdb,
                               const float*  __restrict__ wtok,
                               const int*    __restrict__ idx,
                               const int*    __restrict__ cnt,
                               float* __restrict__ out) {
    __shared__ __bf16 xs[TM][XS_STRIDE];   // 32 tokens x D, padded
    __shared__ __bf16 hs[TM][HS_STRIDE];   // 32 tokens x 128 f-chunk, padded
    __shared__ float  wts[TM];
    __shared__ int    toks[TM];

    const int e    = blockIdx.y;
    const int tile = blockIdx.x;
    const int n_e  = cnt[e];
    if (tile * TM >= n_e) return;

    const int tid  = threadIdx.x;
    const int lane = tid & 31;
    const int wave = tid >> 5;            // 0..7

    if (tid < TM) {
        int p = tile * TM + tid;
        int tok = 0; float w = 0.f;
        if (p < n_e) { tok = idx[e * TT + p]; w = wtok[tok * NE + e]; }
        toks[tid] = tok; wts[tid] = w;
    }
    __syncthreads();

    // ---- stage gathered x rows via Tensor Data Mover: one 1xD 2D descriptor
    //      per token row, so the padded (conflict-free) LDS stride is kept ----
    if (wave == 0) {
        const unsigned lds_base = (unsigned)(size_t)(&xs[0][0]);
        const unsigned long long gbase = (unsigned long long)(size_t)xb;
        for (int r = 0; r < TM; ++r) {
            unsigned long long ga = gbase + (unsigned long long)toks[r] * (DD * 2);
            v4u g0;
            g0[0] = 1u;                                        // count=1
            g0[1] = lds_base + (unsigned)(r * (XS_STRIDE * 2)); // lds_addr
            g0[2] = (unsigned)(ga & 0xFFFFFFFFu);              // global_addr lo
            g0[3] = (unsigned)((ga >> 32) & 0x01FFFFFFu) | (2u << 30); // hi|type=2
            v8i g1;
            g1[0] = (int)(1u << 16);                           // data_size = 2B
            g1[1] = (int)(((unsigned)DD & 0xFFFFu) << 16);     // tensor_dim0 lo16
            g1[2] = (int)((DD >> 16) | (1u << 16));            // dim0 hi | dim1=1
            g1[3] = (int)(((unsigned)DD & 0xFFFFu) << 16);     // tile_dim0 = D
            g1[4] = 1;                                         // tile_dim1 = 1
            g1[5] = DD;                                        // dim0_stride lo32
            g1[6] = 0; g1[7] = 0;
            v4i gz4 = {0, 0, 0, 0};
            v8i gz8 = {0, 0, 0, 0, 0, 0, 0, 0};
            __builtin_amdgcn_tensor_load_to_lds(g0, g1, gz4, gz4, gz8, 0);
        }
        __builtin_amdgcn_s_wait_tensorcnt(0);
    }
    __syncthreads();

    const __bf16* Wge = Wgb + (long long)e * FF * DD;
    const __bf16* Wue = Wub + (long long)e * FF * DD;
    const __bf16* Wde = Wdb + (long long)e * DD * FF;

    const v8f zero = {0.f, 0.f, 0.f, 0.f, 0.f, 0.f, 0.f, 0.f};
    v8f acc0[8], acc1[8];                 // out tiles: 2 m-tiles x 8 n-tiles
#pragma unroll
    for (int i = 0; i < 8; ++i) { acc0[i] = zero; acc1[i] = zero; }

    for (int f0 = 0; f0 < FF; f0 += 128) {
        if (f0 + 128 < FF) {              // L2 prefetch hint for next chunk
            const __bf16* pfg = Wge + (long long)(f0 + 128 + (tid >> 1)) * DD
                                + (tid & 1) * 512;
            const __bf16* pfu = Wue + (long long)(f0 + 128 + (tid >> 1)) * DD
                                + (tid & 1) * 512;
            __builtin_prefetch(pfg, 0, 1);
            __builtin_prefetch(pfu, 0, 1);
        }

        // ---- phase 1: wave w owns f-tile w; gate & up for both m-tiles ----
        const int fb = f0 + wave * 16;
        v8f g0 = zero, g1 = zero, u0 = zero, u1 = zero;
#pragma unroll 4
        for (int k = 0; k < DD; k += 32) {
            v16bf a0 = load_A_frag<XS_STRIDE>(&xs[0][0], k, lane);
            v16bf a1 = load_A_frag<XS_STRIDE>(&xs[16][0], k, lane);
            v16bf bg = load_B_frag(Wge, DD, fb, k, lane);
            v16bf bu = load_B_frag(Wue, DD, fb, k, lane);
            g0 = wmma_bf16(a0, bg, g0);
            g1 = wmma_bf16(a1, bg, g1);
            u0 = wmma_bf16(a0, bu, u0);
            u1 = wmma_bf16(a1, bu, u1);
        }
        {   // silu(g)*u -> bf16 h-chunk in LDS
            const int n  = lane & 15;
            const int mb = (lane >> 4) * 8;
#pragma unroll
            for (int r = 0; r < 8; ++r) {
                float gv0 = g0[r];
                float hv0 = (gv0 / (1.f + __expf(-gv0))) * u0[r];
                hs[mb + r][wave * 16 + n] = (__bf16)hv0;
                float gv1 = g1[r];
                float hv1 = (gv1 / (1.f + __expf(-gv1))) * u1[r];
                hs[16 + mb + r][wave * 16 + n] = (__bf16)hv1;
            }
        }
        __syncthreads();

        // ---- phase 2: wave w owns out cols [w*128, +128); both m-tiles ----
        const int db = wave * 128;
#pragma unroll
        for (int kt = 0; kt < 4; ++kt) {
            v16bf a0 = load_A_frag<HS_STRIDE>(&hs[0][0],  kt * 32, lane);
            v16bf a1 = load_A_frag<HS_STRIDE>(&hs[16][0], kt * 32, lane);
#pragma unroll
            for (int nt = 0; nt < 8; ++nt) {
                v16bf bd = load_B_frag(Wde, FF, db + nt * 16, f0 + kt * 32, lane);
                acc0[nt] = wmma_bf16(a0, bd, acc0[nt]);
                acc1[nt] = wmma_bf16(a1, bd, acc1[nt]);
            }
        }
        __syncthreads();                  // hs reused next chunk
    }

    // ---- epilogue: scale by routing weight; exactly 2 atomic adds/element
    //      across the whole grid -> bit-deterministic ----
    {
        const int n  = lane & 15;
        const int mb = (lane >> 4) * 8;
#pragma unroll
        for (int nt = 0; nt < 8; ++nt) {
            const int dcol = wave * 128 + nt * 16 + n;
#pragma unroll
            for (int r = 0; r < 8; ++r) {
                const int m0 = mb + r, m1 = 16 + mb + r;
                unsafeAtomicAdd(out + (long long)toks[m0] * DD + dcol,
                                acc0[nt][r] * wts[m0]);
                unsafeAtomicAdd(out + (long long)toks[m1] * DD + dcol,
                                acc1[nt][r] * wts[m1]);
            }
        }
    }
}

// ---------------------------------------------------------------------------
extern "C" void kernel_launch(void* const* d_in, const int* in_sizes, int n_in,
                              void* d_out, int out_size, void* d_ws, size_t ws_size,
                              hipStream_t stream) {
    const float* x     = (const float*)d_in[0];
    const float* Wgate = (const float*)d_in[1];
    const float* Wg    = (const float*)d_in[2];
    const float* Wu    = (const float*)d_in[3];
    const float* Wd    = (const float*)d_in[4];
    float* out = (float*)d_out;

    // workspace layout (all 64B-aligned segments):
    //   wtok [T*E] f32 | idx [E*T] i32 | cnt [E] i32 (pad) |
    //   xb [T*D] bf16 | Wgb | Wub | Wdb [E*F*D] bf16 each   (~218.6 MB total)
    char* ws = (char*)d_ws;
    float*  wtok = (float*)ws;                    ws += (size_t)TT * NE * 4;
    int*    idx  = (int*)ws;                      ws += (size_t)NE * TT * 4;
    int*    cnt  = (int*)ws;                      ws += 64;
    __bf16* xb   = (__bf16*)ws;                   ws += (size_t)TT * DD * 2;
    __bf16* Wgb  = (__bf16*)ws;                   ws += (size_t)NE * FF * DD * 2;
    __bf16* Wub  = (__bf16*)ws;                   ws += (size_t)NE * FF * DD * 2;
    __bf16* Wdb  = (__bf16*)ws;

    const int wn4 = (NE * FF * DD) / 4;           // per weight tensor
    const int xn4 = (TT * DD) / 4;
    moe_cvt_kernel<<<(wn4 + 255) / 256, 256, 0, stream>>>((const float4*)Wg, (v4bf*)Wgb, wn4);
    moe_cvt_kernel<<<(wn4 + 255) / 256, 256, 0, stream>>>((const float4*)Wu, (v4bf*)Wub, wn4);
    moe_cvt_kernel<<<(wn4 + 255) / 256, 256, 0, stream>>>((const float4*)Wd, (v4bf*)Wdb, wn4);
    moe_cvt_kernel<<<(xn4 + 255) / 256, 256, 0, stream>>>((const float4*)x, (v4bf*)xb, xn4);

    moe_zero_kernel<<<(xn4 + 255) / 256, 256, 0, stream>>>((float4*)out, xn4);
    moe_router_kernel<<<TT / 8, 256, 0, stream>>>(x, Wgate, wtok);
    moe_compact_kernel<<<NE, 32, 0, stream>>>(wtok, idx, cnt);

    dim3 grid(TT / TM, NE);
    moe_ffn_kernel<<<grid, 256, 0, stream>>>(xb, Wgb, Wub, Wdb,
                                             wtok, idx, cnt, out);
}
